// NonParametricConcordanceHead_90159953477959
// MI455X (gfx1250) — matrix operand order repaired and checked by reading, not compiled, not run on hardware
//
#include <hip/hip_runtime.h>
#include <hip/hip_bf16.h>

typedef __attribute__((ext_vector_type(16))) _Float16 v16h;
typedef __attribute__((ext_vector_type(8)))  float    v8f;
typedef __attribute__((ext_vector_type(4)))  int      v4i;
typedef __attribute__((address_space(1))) v4i as1_v4i;
typedef __attribute__((address_space(3))) v4i as3_v4i;

#define HH 96
#define WW 96
#define DD 768
#define NPIX (HH*WW)      // 9216
#define KOFF 49
#define RR 3
#define TAU_ 0.1f
#define DCHUNKS (DD/32)   // 24
#define XT (WW/16)        // 6

__device__ __forceinline__ int clampi(int v, int lo, int hi){ return v<lo?lo:(v>hi?hi:v); }

// ---------------------------------------------------------------------------
// CDNA5 async global->LDS helpers (guarded; fall back to sync copy)
// ---------------------------------------------------------------------------
#if defined(__gfx1250__) && __has_builtin(__builtin_amdgcn_global_load_async_to_lds_b128)
#define HAVE_ASYNC_LDS 1
#else
#define HAVE_ASYNC_LDS 0
#endif

__device__ __forceinline__ void async_b128(const unsigned* g, unsigned* l)
{
#if HAVE_ASYNC_LDS
    __builtin_amdgcn_global_load_async_to_lds_b128(
        (as1_v4i*)(void*)g,
        (as3_v4i*)(void*)l, 0, 0);
#else
    uint4 v = *(const uint4*)g;
    *(uint4*)l = v;
#endif
}

template<int N>
__device__ __forceinline__ void wait_async_le()
{
#if defined(__gfx1250__)
# if __has_builtin(__builtin_amdgcn_s_wait_asynccnt)
    __builtin_amdgcn_s_wait_asynccnt((unsigned short)N);
# elif HAVE_ASYNC_LDS
    if (N == 0) asm volatile("s_wait_asynccnt 0x0" ::: "memory");
    else        asm volatile("s_wait_asynccnt 0x4" ::: "memory");
# endif
#endif
}

// ---------------------------------------------------------------------------
// 1) per-pixel L2 normalize (over D=768), f32 -> f16, wave-per-pixel
// ---------------------------------------------------------------------------
__global__ void k_normalize(const float* __restrict__ rubin,
                            const float* __restrict__ vis,
                            _Float16* __restrict__ rubinN,
                            _Float16* __restrict__ visN,
                            int totalPix)
{
    int wv   = threadIdx.x >> 5;
    int lane = threadIdx.x & 31;
    int p = blockIdx.x * 8 + wv;
    if (p >= 2*totalPix) return;
    const float* src; _Float16* dst; int pix;
    if (p < totalPix) { src = rubin; dst = rubinN; pix = p; }
    else              { src = vis;   dst = visN;   pix = p - totalPix; }
    const float* v = src + (size_t)pix * DD;
    float x[24];
    float ss = 0.f;
    #pragma unroll
    for (int i = 0; i < 24; ++i) { x[i] = v[lane + i*32]; ss += x[i]*x[i]; }
    #pragma unroll
    for (int off = 16; off > 0; off >>= 1) ss += __shfl_xor(ss, off, 32);
    float inv = 1.f / fmaxf(sqrtf(ss), 1e-6f);
    _Float16* o = dst + (size_t)pix * DD;
    #pragma unroll
    for (int i = 0; i < 24; ++i) o[lane + i*32] = (_Float16)(x[i]*inv);
}

// ---------------------------------------------------------------------------
// 2) banded correlation volume via v_wmma_f32_16x16x32_f16
//    block = 256 threads (8 waves); one block per (b, y, 16-pixel x tile)
//    waves 0..6 each own one dy; 2 WMMA N-tiles cover vis cols [x0-3, x0+28]
//    K-chunks double-buffered in LDS via GLOBAL_LOAD_ASYNC_TO_LDS_B128.
// ---------------------------------------------------------------------------
// Stage one 32-deep K chunk: 1024 b128 segments (64 A + 896 B + 64 pad),
// exactly 4 async instructions per wave -> uniform ASYNCcnt accounting.
__device__ __forceinline__ void stage_chunk(const unsigned* __restrict__ rA,
                                            const unsigned* __restrict__ rB,
                                            unsigned* lAbuf, unsigned* lBbuf,
                                            int b, int y, int x0, int c, int tid)
{
    #pragma unroll
    for (int t = 0; t < 4; ++t) {
        int i = tid + t*256;
        if (i > 959) i = 959;          // pad: duplicate last segment (same data)
        if (i < 64) {
            int pix = i >> 2, seg = i & 3;
            size_t gdw = ((size_t)(b*NPIX + y*WW + x0 + pix)) * (DD/2)
                       + (size_t)c*16 + seg*4;
            async_b128(rA + gdw, lAbuf + i*4);
        } else {
            int j = i - 64;            // 0..895
            int dyi = j >> 7;          // 32 cols * 4 segs = 128 per dy
            int rem = j & 127;
            int col = rem >> 2, seg = rem & 3;
            int yr = clampi(y + dyi - RR, 0, HH-1);
            int xc = clampi(x0 - RR + col, 0, WW-1);
            size_t gdw = ((size_t)(b*NPIX + yr*WW + xc)) * (DD/2)
                       + (size_t)c*16 + seg*4;
            async_b128(rB + gdw, lBbuf + j*4);
        }
    }
}

__global__ void k_corr_wmma(const _Float16* __restrict__ rubinN,
                            const _Float16* __restrict__ visN,
                            float* __restrict__ logits)
{
    __shared__ unsigned lA[2][16*16];     // [buf][pix][dword]
    __shared__ unsigned lB[2][7*32*16];   // [buf][dy][col][dword]
    __shared__ float    lC[7*512];        // per-wave C staging (2 tiles 16x16)

    const int tid  = threadIdx.x;
    const int wv   = tid >> 5;
    const int lane = tid & 31;
    int t  = blockIdx.x;
    int xw = t % XT; t /= XT;
    int y  = t % HH; t /= HH;
    int b  = t;
    int x0 = xw * 16;

    const unsigned* rA = (const unsigned*)rubinN;
    const unsigned* rB = (const unsigned*)visN;

    v8f acc0 = {0.f,0.f,0.f,0.f,0.f,0.f,0.f,0.f};
    v8f acc1 = {0.f,0.f,0.f,0.f,0.f,0.f,0.f,0.f};

    // prologue: stage chunk 0 into buffer 0
    stage_chunk(rA, rB, lA[0], lB[0], b, y, x0, 0, tid);

    for (int c = 0; c < DCHUNKS; ++c) {
        const int cur = c & 1;
        if (c + 1 < DCHUNKS) {
            stage_chunk(rA, rB, lA[cur^1], lB[cur^1], b, y, x0, c+1, tid);
            wait_async_le<4>();       // group(c) done; group(c+1) may be in flight
        } else {
            wait_async_le<0>();
        }
        __syncthreads();              // chunk c visible to all waves

        if (wv < 7) {
            const int g = lane >> 4;   // half-wave group
            const int m = lane & 15;   // row (A) / col (B)
            union { v16h v; unsigned u[8]; } Af, Bf;
            // A fragment (this code's contiguous k order, same as B): dword kd = v + 8g
            #pragma unroll
            for (int v = 0; v < 8; ++v) {
                int kdA = v + 8*g;
                Af.u[v] = lA[cur][m*16 + kdA];
            }
            #pragma unroll
            for (int nt = 0; nt < 2; ++nt) {
                int col = nt*16 + m;
                // B fragment (this code's contiguous k order, same as A): lanes 0-15 K=0..15, 16-31 K=16..31
                #pragma unroll
                for (int v = 0; v < 8; ++v) {
                    int kdB = v + 8*g;
                    Bf.u[v] = lB[cur][(wv*32 + col)*16 + kdB];
                }
                if (nt == 0)
                    acc0 = __builtin_amdgcn_wmma_f32_16x16x32_f16(
                        false, Af.v, false, Bf.v, (short)0, acc0, false, false);
                else
                    acc1 = __builtin_amdgcn_wmma_f32_16x16x32_f16(
                        false, Af.v, false, Bf.v, (short)0, acc1, false, false);
            }
        }
        __syncthreads();              // chunk c fully consumed before buffer reuse
    }

    // extract diagonal band: logits[b, k=dy*7+dx, y, x0+m] = C[m, m+dx] / sqrt(D)
    if (wv < 7) {
        const int g = lane >> 4;
        const int n = lane & 15;
        float* cw = lC + wv*512;
        #pragma unroll
        for (int v = 0; v < 8; ++v) {
            cw[0*256 + (v + 8*g)*16 + n] = acc0[v];
            cw[1*256 + (v + 8*g)*16 + n] = acc1[v];
        }
        const float scale = 0.03608439182435161f; // 1/sqrt(768)
        for (int idx = lane; idx < 112; idx += 32) {
            int m   = idx / 7;
            int dxi = idx % 7;
            int nc  = m + dxi;                 // 0..21 into the 32-col window
            float val = cw[(nc >> 4)*256 + m*16 + (nc & 15)] * scale;
            int k = wv*7 + dxi;
            logits[((size_t)b*KOFF + k)*NPIX + y*WW + x0 + m] = val;
        }
    }
}

// ---------------------------------------------------------------------------
// 3) mean of logits over H*W per (b,k)
// ---------------------------------------------------------------------------
__global__ void k_meanlogits(const float* __restrict__ logits, float* __restrict__ meanL)
{
    __shared__ float s[256];
    int bk = blockIdx.x;
    const float* p = logits + (size_t)bk * NPIX;
    float ss = 0.f;
    for (int i = threadIdx.x; i < NPIX; i += 256) ss += p[i];
    s[threadIdx.x] = ss; __syncthreads();
    for (int st = 128; st > 0; st >>= 1) {
        if (threadIdx.x < st) s[threadIdx.x] += s[threadIdx.x + st];
        __syncthreads();
    }
    if (threadIdx.x == 0) meanL[bk] = s[0] * (1.f/(float)NPIX);
}

// ---------------------------------------------------------------------------
// 4) global softmax over K per batch -> dy_global, dx_global, conf_global
// ---------------------------------------------------------------------------
__global__ void k_globalsoftmax(const float* __restrict__ meanL,
                                float* __restrict__ dyg, float* __restrict__ dxg,
                                float* __restrict__ out_confg)
{
    __shared__ float sl[KOFF];
    int b = blockIdx.x;
    if (threadIdx.x < KOFF) sl[threadIdx.x] = meanL[b*KOFF + threadIdx.x];
    __syncthreads();
    if (threadIdx.x == 0) {
        float mx = -1e30f;
        for (int k = 0; k < KOFF; ++k) mx = fmaxf(mx, sl[k]);
        float sum = 0.f, dys = 0.f, dxs = 0.f, pmax = 0.f;
        for (int k = 0; k < KOFF; ++k) {
            float e = __expf((sl[k]-mx) * (1.f/TAU_));
            sum += e;
            dys += e * (float)(k/7 - RR);
            dxs += e * (float)(k%7 - RR);
            pmax = fmaxf(pmax, e);
        }
        float inv = 1.f/sum;
        dyg[b] = dys*inv; dxg[b] = dxs*inv; out_confg[b] = pmax*inv;
    }
}

// ---------------------------------------------------------------------------
// 5) local softmax + blend
// ---------------------------------------------------------------------------
__global__ void k_localsoftmax(const float* __restrict__ logits,
                               const float* __restrict__ dyg, const float* __restrict__ dxg,
                               float* __restrict__ dyt, float* __restrict__ dxt,
                               float* __restrict__ out_conf, float* __restrict__ out_lw,
                               int total)
{
    int pix = blockIdx.x*256 + threadIdx.x;
    if (pix >= total) return;
    int b  = pix / NPIX;
    int sp = pix % NPIX;
    const float* p = logits + (size_t)b*KOFF*NPIX + sp;
    float l[KOFF];
    float mx = -1e30f;
    #pragma unroll
    for (int k = 0; k < KOFF; ++k) { l[k] = p[(size_t)k*NPIX]; mx = fmaxf(mx, l[k]); }
    float sum = 0.f, dys = 0.f, dxs = 0.f, pmax = 0.f;
    #pragma unroll
    for (int k = 0; k < KOFF; ++k) {
        float e = __expf((l[k]-mx)*(1.f/TAU_));
        sum += e;
        dys += e*(float)(k/7 - RR);
        dxs += e*(float)(k%7 - RR);
        pmax = fmaxf(pmax, e);
    }
    float inv  = 1.f/sum;
    float conf = pmax*inv;
    const float u = 1.f/(float)KOFF;
    float lw = (conf - u)/(1.f - u);
    lw = fminf(fmaxf(lw, 0.f), 1.f);
    dyt[pix] = lw*(dys*inv) + (1.f-lw)*dyg[b];
    dxt[pix] = lw*(dxs*inv) + (1.f-lw)*dxg[b];
    out_conf[pix] = conf;
    out_lw[pix]   = lw;
}

// ---------------------------------------------------------------------------
// 6) 3x3 zero-pad avg pool + sky scaling
// ---------------------------------------------------------------------------
__global__ void k_pool(const float* __restrict__ dyt, const float* __restrict__ dxt,
                       float* __restrict__ dra, float* __restrict__ ddec,
                       const int* pH, const int* pW, const int* pHv, const int* pWv,
                       int total)
{
    int pix = blockIdx.x*256 + threadIdx.x;
    if (pix >= total) return;
    int b  = pix / NPIX;
    int sp = pix % NPIX;
    int y = sp / WW, x = sp % WW;
    float sy = 0.f, sx = 0.f;
    #pragma unroll
    for (int dy = -1; dy <= 1; ++dy)
      #pragma unroll
      for (int dx = -1; dx <= 1; ++dx) {
        int yy = y+dy, xx = x+dx;
        if (yy >= 0 && yy < HH && xx >= 0 && xx < WW) {
            int q = b*NPIX + yy*WW + xx;
            sy += dyt[q]; sx += dxt[q];
        }
      }
    float sky_y = (float)pHv[0]*0.1f/(float)pH[0];
    float sky_x = (float)pWv[0]*0.1f/(float)pW[0];
    ddec[pix] = (sy*(1.f/9.f))*sky_y;
    dra[pix]  = (sx*(1.f/9.f))*sky_x;
}

// ---------------------------------------------------------------------------
extern "C" void kernel_launch(void* const* d_in, const int* in_sizes, int n_in,
                              void* d_out, int out_size, void* d_ws, size_t ws_size,
                              hipStream_t stream)
{
    const float* rubin = (const float*)d_in[0];
    const float* vis   = (const float*)d_in[1];
    const int* pH  = (const int*)d_in[2];
    const int* pW  = (const int*)d_in[3];
    const int* pHv = (const int*)d_in[4];
    const int* pWv = (const int*)d_in[5];

    const int B = in_sizes[0] / (NPIX*DD);
    const int totalPix = B*NPIX;

    char* ws = (char*)d_ws;
    size_t off = 0;
    _Float16* rubinN = (_Float16*)(ws + off); off += (size_t)totalPix*DD*2;
    _Float16* visN   = (_Float16*)(ws + off); off += (size_t)totalPix*DD*2;
    float* logits = (float*)(ws + off); off += (size_t)B*KOFF*NPIX*4;
    float* meanL  = (float*)(ws + off); off += (size_t)B*KOFF*4;
    float* dyg    = (float*)(ws + off); off += (size_t)B*4;
    float* dxg    = (float*)(ws + off); off += (size_t)B*4;
    float* dyt    = (float*)(ws + off); off += (size_t)totalPix*4;
    float* dxt    = (float*)(ws + off); off += (size_t)totalPix*4;
    (void)ws_size; (void)n_in; (void)out_size;

    float* out     = (float*)d_out;
    float* o_dra   = out;
    float* o_ddec  = out + totalPix;
    float* o_conf  = out + 2*totalPix;
    float* o_confg = out + 3*totalPix;
    float* o_lw    = out + 3*totalPix + B;

    k_normalize<<<(2*totalPix + 7)/8, 256, 0, stream>>>(rubin, vis, rubinN, visN, totalPix);
    k_corr_wmma<<<B*HH*XT, 256, 0, stream>>>(rubinN, visN, logits);
    k_meanlogits<<<B*KOFF, 256, 0, stream>>>(logits, meanL);
    k_globalsoftmax<<<B, 64, 0, stream>>>(meanL, dyg, dxg, o_confg);
    k_localsoftmax<<<(totalPix + 255)/256, 256, 0, stream>>>(logits, dyg, dxg, dyt, dxt, o_conf, o_lw, totalPix);
    k_pool<<<(totalPix + 255)/256, 256, 0, stream>>>(dyt, dxt, o_dra, o_ddec, pH, pW, pHv, pWv, totalPix);
}
